// DenseMatcher_67388036874486
// MI455X (gfx1250) — matrix operand order repaired and checked by reading, not compiled, not run on hardware
//
#include <hip/hip_runtime.h>
#include <hip/hip_bf16.h>

#define NPIX 9216   // H*W
#define CC   256
#define WDIM 96
#define HDIM 96
#define THRESH 0.9f

typedef __attribute__((ext_vector_type(16))) __bf16 v16bf;
typedef __attribute__((ext_vector_type(8)))  float  v8f;
typedef int v4i_vs __attribute__((vector_size(16)));   // builtin's element type

__device__ __forceinline__ unsigned short f2bf(float f) {
    unsigned u = __builtin_bit_cast(unsigned, f);
    unsigned r = (u + 0x7FFFu + ((u >> 16) & 1u)) >> 16;   // RNE
    return (unsigned short)r;
}

// ---- CDNA5 async-copy helpers (ASYNCcnt path) --------------------------------
__device__ __forceinline__ unsigned lds_addr32(const void* p) {
    return (unsigned)(unsigned long long)
        (const __attribute__((address_space(3))) void*)p;
}

__device__ __forceinline__ void async_b128(void* lds, const void* g) {
#if __has_builtin(__builtin_amdgcn_global_load_async_to_lds_b128)
    __builtin_amdgcn_global_load_async_to_lds_b128(
        (__attribute__((address_space(1))) v4i_vs*)g,
        (__attribute__((address_space(3))) v4i_vs*)lds, 0, 0);
#else
    asm volatile("global_load_async_to_lds_b128 %0, %1, off"
                 :: "v"(lds_addr32(lds)), "v"((unsigned long long)g)
                 : "memory");
#endif
}

template <int N>
__device__ __forceinline__ void wait_async() {
#if __has_builtin(__builtin_amdgcn_s_wait_asynccnt)
    __builtin_amdgcn_s_wait_asynccnt(N);
#else
    asm volatile("s_wait_asynccnt %0" :: "n"(N));
#endif
}

// Phase 1: per-pixel L2 normalize over channels, emit bf16 transposed [pixel][channel]
__global__ __launch_bounds__(256) void normalize_pack(const float* __restrict__ src,
                                                      unsigned short* __restrict__ dst) {
    int n = blockIdx.x * blockDim.x + threadIdx.x;
    if (n >= NPIX) return;
    float s = 0.f;
    for (int c = 0; c < CC; ++c) {
        float v = src[c * NPIX + n];
        s += v * v;
    }
    float inv = rsqrtf(s);
    for (int c = 0; c < CC; c += 8) {
        unsigned short t[8];
#pragma unroll
        for (int i = 0; i < 8; ++i) t[i] = f2bf(src[(c + i) * NPIX + n] * inv);
        uint4 pk;
        pk.x = (unsigned)t[0] | ((unsigned)t[1] << 16);
        pk.y = (unsigned)t[2] | ((unsigned)t[3] << 16);
        pk.z = (unsigned)t[4] | ((unsigned)t[5] << 16);
        pk.w = (unsigned)t[6] | ((unsigned)t[7] << 16);
        *(uint4*)&dst[n * CC + c] = pk;
    }
}

// Phase 2: sim = Qt * Pt^T via v_wmma_f32_16x16x32_bf16. 128x128 tile per block,
// 8 waves, each wave 32(n) x 64(m) = 2x4 WMMA tiles. Double-buffered LDS filled
// by global_load_async_to_lds_b128 (ASYNCcnt), rows padded to 80B (bank-clean).
#define TILE  128
#define KSTEP 32
#define LROW  40   // bf16 elements per LDS row (32 data + 8 pad) = 80 bytes

__global__ __launch_bounds__(256) void sim_gemm(const unsigned short* __restrict__ Qt,
                                                const unsigned short* __restrict__ Pt,
                                                float* __restrict__ sim) {
    __shared__ __align__(16) unsigned short lsA[2][TILE * LROW];
    __shared__ __align__(16) unsigned short lsB[2][TILE * LROW];

    const int tid  = threadIdx.x;
    const int lane = tid & 31;
    const int wave = tid >> 5;
    const int nb = blockIdx.y * TILE;
    const int mb = blockIdx.x * TILE;
    const int wn = (wave & 3) * 32;   // wave's n offset inside tile
    const int wm = (wave >> 2) * 64;  // wave's m offset inside tile
    const int lrow = lane & 15;
    const int hi   = lane >> 4;       // half-wave select

    v8f acc[2][4] = {};

    const int crow  = tid >> 1;         // 0..127 : tile row this thread DMAs
    const int chalf = (tid & 1) * 16;   // element offset 0 or 16 (32B chunk)

    // issue this thread's 4 async b128 DMAs for k-chunk kb into buffer buf
    auto issue = [&](int buf, int kb) {
        const unsigned short* ga = &Qt[(nb + crow) * CC + kb + chalf];
        const unsigned short* gb = &Pt[(mb + crow) * CC + kb + chalf];
        async_b128(&lsA[buf][crow * LROW + chalf],     ga);
        async_b128(&lsA[buf][crow * LROW + chalf + 8], ga + 8);
        async_b128(&lsB[buf][crow * LROW + chalf],     gb);
        async_b128(&lsB[buf][crow * LROW + chalf + 8], gb + 8);
    };

    issue(0, 0);

    for (int kb = 0, it = 0; kb < CC; kb += KSTEP, ++it) {
        const int  cur  = it & 1;
        const bool more = (kb + KSTEP) < CC;
        if (more) issue(cur ^ 1, kb + KSTEP);   // overlap DMA of next chunk

        // in-order async completion: <=4 outstanding means current buffer landed
        if (more) wait_async<4>(); else wait_async<0>();
        __syncthreads();                         // all waves see buffer filled

        // ---- fragment loads (contiguous 16B chunks per CDNA5 WMMA layouts)
        union Frag { v16bf v; uint4 u[2]; };
        Frag a[2], b[4];
        const int akb = hi ? 8 : 0;   // A: lanes 16-31 hold K=8..15 / 24..31
#pragma unroll
        for (int t = 0; t < 2; ++t) {
            const unsigned short* base = &lsA[cur][(wn + t * 16 + lrow) * LROW];
            a[t].u[0] = *(const uint4*)&base[akb];
            a[t].u[1] = *(const uint4*)&base[akb + 16];
        }
        const int bkb = hi ? 8 : 0;   // B: lanes 16-31 hold K=8..15 / 24..31
#pragma unroll
        for (int t = 0; t < 4; ++t) {
            const unsigned short* base = &lsB[cur][(wm + t * 16 + lrow) * LROW];
            b[t].u[0] = *(const uint4*)&base[bkb];
            b[t].u[1] = *(const uint4*)&base[bkb + 16];
        }

#pragma unroll
        for (int i = 0; i < 2; ++i)
#pragma unroll
            for (int j = 0; j < 4; ++j)
                acc[i][j] = __builtin_amdgcn_wmma_f32_16x16x32_bf16(
                    false, a[i].v, false, b[j].v, (short)0, acc[i][j], false, false);

        __syncthreads();   // all fragment reads done before buffer is re-filled
    }

    // ---- epilogue: C/D layout -> VGPR r: lanes0-15 M=r, lanes16-31 M=8+r; N=lane&15
#pragma unroll
    for (int i = 0; i < 2; ++i) {
#pragma unroll
        for (int j = 0; j < 4; ++j) {
            int m = mb + wm + j * 16 + lrow;
#pragma unroll
            for (int r = 0; r < 8; ++r) {
                int n = nb + wn + i * 16 + hi * 8 + r;
                sim[(long)n * NPIX + m] = acc[i][j][r];
            }
        }
    }
}

// Phase 3a: argmax over rows n for each column j (coalesced across j)
__global__ __launch_bounds__(256) void col_argmax(const float* __restrict__ sim,
                                                  int* __restrict__ colmax) {
    int j = blockIdx.x * blockDim.x + threadIdx.x;
    if (j >= NPIX) return;
    float best = -1e30f; int bi = 0;
    for (int n = 0; n < NPIX; ++n) {
        float v = sim[(long)n * NPIX + j];
        if (v > best) { best = v; bi = n; }   // first-max like jnp.argmax
    }
    colmax[j] = bi;
}

// Phase 3b: block-per-row argmax over columns m
__global__ __launch_bounds__(256) void row_argmax(const float* __restrict__ sim,
                                                  int* __restrict__ rowmax) {
    __shared__ float sv[256];
    __shared__ int   si[256];
    int n = blockIdx.x;
    const float* row = sim + (long)n * NPIX;
    float best = -1e30f; int bi = NPIX;
    for (int m = threadIdx.x; m < NPIX; m += 256) {
        float v = row[m];
        if (v > best) { best = v; bi = m; }
    }
    sv[threadIdx.x] = best; si[threadIdx.x] = bi;
    __syncthreads();
    for (int s = 128; s > 0; s >>= 1) {
        if (threadIdx.x < (unsigned)s) {
            float ov = sv[threadIdx.x + s]; int oi = si[threadIdx.x + s];
            if (ov > sv[threadIdx.x] ||
                (ov == sv[threadIdx.x] && oi < si[threadIdx.x])) {
                sv[threadIdx.x] = ov; si[threadIdx.x] = oi;
            }
        }
        __syncthreads();
    }
    if (threadIdx.x == 0) rowmax[n] = si[0];
}

// Phase 3c: mutual-NN + threshold + index math; append after sim in d_out
__global__ __launch_bounds__(256) void finalize(const float* __restrict__ sim,
                                                const int* __restrict__ rowmax,
                                                const int* __restrict__ colmax,
                                                float* __restrict__ out) {
    int j = blockIdx.x * blockDim.x + threadIdx.x;
    if (j >= NPIX) return;
    int qi = colmax[j];
    float sv = sim[(long)qi * NPIX + j];
    bool mutual = (rowmax[qi] == j) && (sv > THRESH);
    int qd = qi / WDIM, qm = qi % WDIM;
    int pd = j  / WDIM, pm = j  % WDIM;
    bool valid = mutual && (qd + 1 < HDIM) && (pd + 1 < HDIM);
    long base = (long)NPIX * NPIX;
    out[base + 0L * NPIX + j] = valid ? 1.f : 0.f;
    out[base + 1L * NPIX + j] = (float)qi;
    out[base + 2L * NPIX + j] = (float)qd;
    out[base + 3L * NPIX + j] = (float)qm;
    out[base + 4L * NPIX + j] = (float)pd;
    out[base + 5L * NPIX + j] = (float)pm;
}

extern "C" void kernel_launch(void* const* d_in, const int* in_sizes, int n_in,
                              void* d_out, int out_size, void* d_ws, size_t ws_size,
                              hipStream_t stream) {
    const float* q = (const float*)d_in[0];
    const float* p = (const float*)d_in[1];
    float* out = (float*)d_out;

    char* ws = (char*)d_ws;
    const size_t packBytes = (size_t)NPIX * CC * sizeof(unsigned short); // 4.72 MB
    unsigned short* Qt = (unsigned short*)ws;
    unsigned short* Pt = (unsigned short*)(ws + packBytes);
    int* rowmax = (int*)(ws + 2 * packBytes);
    int* colmax = rowmax + NPIX;

    normalize_pack<<<NPIX / 256, 256, 0, stream>>>(q, Qt);
    normalize_pack<<<NPIX / 256, 256, 0, stream>>>(p, Pt);

    dim3 grid(NPIX / TILE, NPIX / TILE);   // 72 x 72
    sim_gemm<<<grid, 256, 0, stream>>>(Qt, Pt, out);

    col_argmax<<<NPIX / 256, 256, 0, stream>>>(out, colmax);
    row_argmax<<<NPIX, 256, 0, stream>>>(out, rowmax);
    finalize<<<NPIX / 256, 256, 0, stream>>>(out, rowmax, colmax, out);
}
